// MultiheadAttention_83116207112582
// MI455X (gfx1250) — matrix operand order repaired and checked by reading, not compiled, not run on hardware
//
#include <hip/hip_runtime.h>

// MI455X / gfx1250 multi-head attention with prefix KV.
// All GEMM work routed through v_wmma_f32_16x16x32_bf16 (wave32 WMMA),
// softmax kept in fp32 VALU, online (flash) softmax over M=2064 keys.
// fp32 operands are converted to bf16 ONCE into workspace; hot loops are
// software-pipelined so global_load_b128 overlaps v_wmma (partial loadcnt
// waits instead of draining to zero before every matrix op).

typedef __attribute__((ext_vector_type(16))) __bf16 v16bf;
typedef __attribute__((ext_vector_type(8)))  float  v8f;

union Frag {
    v16bf          v;
    unsigned int   u[8];
    unsigned short s[16];
    uint4          q[2];
};

__device__ __forceinline__ unsigned short f2bf(float f) {
    unsigned int u = __float_as_uint(f);
    unsigned int r = u + 0x7FFFu + ((u >> 16) & 1u);   // round-to-nearest-even
    return (unsigned short)(r >> 16);
}

// Problem constants
#define NQ    2048
#define BB    4
#define CC    768
#define HH    12
#define HD    64
#define PP    16
#define MREAL 2064          // P + N
#define MPAD  2080          // padded to multiple of 32

// ---------------------------------------------------------------------------
// Kernel 0: one-shot fp32 -> bf16 conversion (8 elements / thread).
// ---------------------------------------------------------------------------
__global__ __launch_bounds__(256)
void cvt_bf16_kernel(const float* __restrict__ src,
                     unsigned short* __restrict__ dst, int n8)
{
    int i = blockIdx.x * 256 + threadIdx.x;
    if (i >= n8) return;
    const float4* s4 = (const float4*)src + (size_t)i * 2;
    float4 a = s4[0];
    float4 b = s4[1];
    uint4 o;
    o.x = ((unsigned)f2bf(a.y) << 16) | f2bf(a.x);
    o.y = ((unsigned)f2bf(a.w) << 16) | f2bf(a.z);
    o.z = ((unsigned)f2bf(b.y) << 16) | f2bf(b.x);
    o.w = ((unsigned)f2bf(b.w) << 16) | f2bf(b.z);
    ((uint4*)dst)[i] = o;
}

// ---------------------------------------------------------------------------
// Kernel 1: fused QKV projection.  (8192 x 768) x (768 x 2304) in bf16 WMMA.
// Each wave: 4 row-tiles x 1 col-tile (W fragment reused 4x), double-buffered
// fragments so next k-step loads overlap current WMMAs.
// Writes Q (b,h,n,d), K (b,h,m,d) at m = P+n, and V transposed (b,h,d,m).
// ---------------------------------------------------------------------------
__global__ __launch_bounds__(256)
void qkv_proj_kernel(const unsigned short* __restrict__ xb,
                     const unsigned short* __restrict__ wb,
                     unsigned short* __restrict__ qbuf,
                     unsigned short* __restrict__ kbuf,
                     unsigned short* __restrict__ vtbuf)
{
    const int lane = threadIdx.x & 31;
    const int wave = threadIdx.x >> 5;
    const int gw   = blockIdx.x * 8 + wave;
    const int ROWG = (NQ * BB) / 64;            // 128 row groups of 64 rows
    const int colT = gw / ROWG;                 // 0..143
    const int rowG = gw % ROWG;
    if (colT >= (3 * CC) / 16) return;
    const int o0 = colT * 16;
    const int r0 = rowG * 64;
    const int ln = lane & 15;
    const int hi = lane >> 4;

    const unsigned short* wrow = wb + (size_t)(o0 + ln) * CC + hi * 16;
    const unsigned short* xrow = xb + (size_t)(r0 + ln) * CC + hi * 16;

    v8f acc[4];
#pragma unroll
    for (int rt = 0; rt < 4; ++rt) acc[rt] = (v8f)0.0f;

    Frag bf[2];
    Frag af[2][4];

    auto load_stage = [&](int s, int c0) {
        const unsigned short* wp = wrow + c0;
        bf[s].q[0] = *(const uint4*)(wp);
        bf[s].q[1] = *(const uint4*)(wp + 8);
#pragma unroll
        for (int rt = 0; rt < 4; ++rt) {
            const unsigned short* xp = xrow + (size_t)rt * 16 * CC + c0;
            // A fragment: one contiguous 32B run, k=16h..16h+15
            af[s][rt].q[0] = *(const uint4*)(xp);
            af[s][rt].q[1] = *(const uint4*)(xp + 8);
        }
    };

    const int NSTEP = CC / 32;                  // 24
    load_stage(0, 0);
    for (int s = 0; s < NSTEP; ++s) {
        int cur = s & 1, nxt = cur ^ 1;
        int c0n = (s + 1 < NSTEP) ? (s + 1) * 32 : 0;   // last iter: redundant
        load_stage(nxt, c0n);                            // issue BEFORE WMMAs
#pragma unroll
        for (int rt = 0; rt < 4; ++rt)
            acc[rt] = __builtin_amdgcn_wmma_f32_16x16x32_bf16(
                false, af[cur][rt].v, false, bf[cur].v, (short)0, acc[rt],
                false, false);
    }

#pragma unroll
    for (int rt = 0; rt < 4; ++rt) {
#pragma unroll
        for (int i = 0; i < 8; ++i) {
            int trow = r0 + rt * 16 + (hi ? 8 + i : i);   // token index n*B+b
            int o    = o0 + ln;
            int n    = trow >> 2;                          // /B
            int b    = trow & 3;
            int part = o / CC;
            int oo   = o % CC;
            int h    = oo >> 6;
            int d    = oo & 63;
            int bh   = b * HH + h;
            unsigned short bv = f2bf(acc[rt][i]);
            if (part == 0)
                qbuf[((size_t)bh * NQ + n) * HD + d] = bv;
            else if (part == 1)
                kbuf[((size_t)bh * MPAD + PP + n) * HD + d] = bv;
            else
                vtbuf[((size_t)bh * HD + d) * MPAD + PP + n] = bv;
        }
    }
}

// ---------------------------------------------------------------------------
// Kernel 1b: prefix KV conversion + zero padding of the tail (m in [2064,2080)).
// ---------------------------------------------------------------------------
__global__ __launch_bounds__(256)
void prefix_pad_kernel(const float* __restrict__ pk, const float* __restrict__ pv,
                       unsigned short* __restrict__ kbuf,
                       unsigned short* __restrict__ vtbuf)
{
    int idx = blockIdx.x * 256 + threadIdx.x;          // 48*64*32 work items
    if (idx >= BB * HH * HD * 32) return;
    int j  = idx & 31;
    int d  = (idx >> 5) & 63;
    int bh = idx >> 11;
    int b  = bh / HH, h = bh % HH;
    unsigned short kv = 0, vv = 0;
    int m;
    if (j < PP) {
        m  = j;
        kv = f2bf(pk[((size_t)b * PP + m) * CC + h * HD + d]);
        vv = f2bf(pv[((size_t)b * PP + m) * CC + h * HD + d]);
    } else {
        m = MREAL + (j - PP);                          // zero pad
    }
    kbuf[((size_t)bh * MPAD + m) * HD + d]  = kv;
    vtbuf[((size_t)bh * HD + d) * MPAD + m] = vv;
}

// ---------------------------------------------------------------------------
// Kernel 2: flash attention.  One wave = 16 query rows; 8 waves/block.
// Per 32-key chunk: all K and V fragments are issued up front, so QK^T only
// partially waits (V still in flight) and V completes under the softmax VALU.
// K/V for one head = 0.5 MB -> all heads resident in the 192 MB L2.
// ---------------------------------------------------------------------------
__global__ __launch_bounds__(256)
void attn_kernel(const unsigned short* __restrict__ qbuf,
                 const unsigned short* __restrict__ kbuf,
                 const unsigned short* __restrict__ vtbuf,
                 unsigned short* __restrict__ aout)
{
    __shared__ unsigned short ptile[8][16 * 32];       // per-wave P staging
    const int lane = threadIdx.x & 31;
    const int wave = threadIdx.x >> 5;
    const int bh   = blockIdx.x >> 4;                  // 48 (b,h) pairs
    const int rb   = blockIdx.x & 15;
    const int q0   = rb * 128 + wave * 16;
    const int ln   = lane & 15;
    const int hi   = lane >> 4;

    const unsigned short* qp = qbuf + ((size_t)bh * NQ + q0) * HD;
    const unsigned short* kp = kbuf + (size_t)bh * MPAD * HD;
    const unsigned short* vp = vtbuf + (size_t)bh * HD * MPAD;

    // Q A-fragments (16 x 64 => two 16x32 fragments), live for whole loop.
    Frag qa[2];
#pragma unroll
    for (int hh = 0; hh < 2; ++hh) {
        const unsigned short* src = qp + ln * HD + hh * 32 + hi * 16;
        qa[hh].q[0] = *(const uint4*)(src);
        qa[hh].q[1] = *(const uint4*)(src + 8);
    }

    v8f O[4];
#pragma unroll
    for (int dt = 0; dt < 4; ++dt) O[dt] = (v8f)0.0f;
    float rmax[8], rsum[8];
#pragma unroll
    for (int i = 0; i < 8; ++i) { rmax[i] = -1e30f; rsum[i] = 0.0f; }

    for (int m0 = 0; m0 < MPAD; m0 += 32) {
        // ---- issue ALL chunk loads first: K fragments then V fragments ----
        Frag kb[4];                                    // [t*2+hh]
#pragma unroll
        for (int t = 0; t < 2; ++t)
#pragma unroll
            for (int hh = 0; hh < 2; ++hh) {
                const unsigned short* src =
                    kp + (size_t)(m0 + t * 16 + ln) * HD + hh * 32 + hi * 16;
                kb[t * 2 + hh].q[0] = *(const uint4*)(src);
                kb[t * 2 + hh].q[1] = *(const uint4*)(src + 8);
            }
        Frag vb[4];
#pragma unroll
        for (int dt = 0; dt < 4; ++dt) {
            const unsigned short* src =
                vp + (size_t)(dt * 16 + ln) * MPAD + m0 + hi * 16;
            vb[dt].q[0] = *(const uint4*)(src);
            vb[dt].q[1] = *(const uint4*)(src + 8);
        }
        // ---- scores: S (16q x 32m), two 16-col tiles ----
        v8f S[2];
        S[0] = (v8f)0.0f; S[1] = (v8f)0.0f;
#pragma unroll
        for (int t = 0; t < 2; ++t)
#pragma unroll
            for (int hh = 0; hh < 2; ++hh)
                S[t] = __builtin_amdgcn_wmma_f32_16x16x32_bf16(
                    false, qa[hh].v, false, kb[t * 2 + hh].v, (short)0, S[t],
                    false, false);
        // ---- scale + padding mask ----
#pragma unroll
        for (int t = 0; t < 2; ++t) {
            int  mcol  = m0 + t * 16 + ln;
            bool valid = mcol < MREAL;
#pragma unroll
            for (int i = 0; i < 8; ++i) {
                float s = S[t][i] * 0.125f;            // hd^-0.5 = 1/8
                S[t][i] = valid ? s : -1e30f;
            }
        }
        // ---- online softmax: row max, correction ----
        float corr[8];
#pragma unroll
        for (int i = 0; i < 8; ++i) {
            float cm = fmaxf(S[0][i], S[1][i]);
#pragma unroll
            for (int off = 8; off >= 1; off >>= 1)
                cm = fmaxf(cm, __shfl_xor(cm, off, 32));
            float nm = fmaxf(rmax[i], cm);
            corr[i]  = __expf(rmax[i] - nm);
            rmax[i]  = nm;
        }
#pragma unroll
        for (int t = 0; t < 2; ++t)
#pragma unroll
            for (int i = 0; i < 8; ++i)
                S[t][i] = __expf(S[t][i] - rmax[i]);
#pragma unroll
        for (int i = 0; i < 8; ++i) {
            float ps = S[0][i] + S[1][i];
#pragma unroll
            for (int off = 8; off >= 1; off >>= 1)
                ps += __shfl_xor(ps, off, 32);
            rsum[i] = rsum[i] * corr[i] + ps;
        }
#pragma unroll
        for (int dt = 0; dt < 4; ++dt)
#pragma unroll
            for (int i = 0; i < 8; ++i)
                O[dt][i] *= corr[i];
        // ---- P: C-fragment layout -> A-fragment layout via LDS ----
#pragma unroll
        for (int t = 0; t < 2; ++t) {
            int col = t * 16 + ln;
#pragma unroll
            for (int i = 0; i < 8; ++i) {
                int row = hi ? 8 + i : i;
                ptile[wave][row * 32 + col] = f2bf(S[t][i]);
            }
        }
        asm volatile("s_wait_dscnt 0" ::: "memory");   // same-wave LDS RAW
        Frag pa;
        {
            const unsigned short* src = &ptile[wave][ln * 32 + hi * 16];
            pa.q[0] = *(const uint4*)(src);
            pa.q[1] = *(const uint4*)(src + 8);
        }
        // ---- O += P x V (4 d-tiles); V fragments already resident ----
#pragma unroll
        for (int dt = 0; dt < 4; ++dt)
            O[dt] = __builtin_amdgcn_wmma_f32_16x16x32_bf16(
                false, pa.v, false, vb[dt].v, (short)0, O[dt], false, false);
    }

    // ---- normalize and store, token-major (n*B+b, C) for the out-proj ----
    int h = bh % HH;
    int b = bh / HH;
#pragma unroll
    for (int dt = 0; dt < 4; ++dt) {
#pragma unroll
        for (int i = 0; i < 8; ++i) {
            int qrow = q0 + (hi ? 8 + i : i);
            int d    = dt * 16 + ln;
            float v  = O[dt][i] / rsum[i];
            aout[((size_t)qrow * BB + b) * CC + h * HD + d] = f2bf(v);
        }
    }
}

// ---------------------------------------------------------------------------
// Kernel 3: output projection (8192 x 768) x (768 x 768) + bias, fp32 out.
// Both operands already bf16 in workspace; double-buffered like kernel 1.
// ---------------------------------------------------------------------------
__global__ __launch_bounds__(256)
void out_proj_kernel(const unsigned short* __restrict__ att,
                     const unsigned short* __restrict__ wb,
                     const float* __restrict__ bias,
                     float* __restrict__ out)
{
    const int lane = threadIdx.x & 31;
    const int wave = threadIdx.x >> 5;
    const int gw   = blockIdx.x * 8 + wave;
    const int ROWG = (NQ * BB) / 64;                   // 128
    const int colT = gw / ROWG;                        // 0..47
    const int rowG = gw % ROWG;
    if (colT >= CC / 16) return;
    const int o0 = colT * 16;
    const int r0 = rowG * 64;
    const int ln = lane & 15;
    const int hi = lane >> 4;

    const unsigned short* wrow = wb + (size_t)(o0 + ln) * CC + hi * 16;
    const unsigned short* xrow = att + (size_t)(r0 + ln) * CC + hi * 16;

    v8f acc[4];
#pragma unroll
    for (int rt = 0; rt < 4; ++rt) acc[rt] = (v8f)0.0f;

    Frag bf[2];
    Frag af[2][4];

    auto load_stage = [&](int s, int c0) {
        const unsigned short* wp = wrow + c0;
        bf[s].q[0] = *(const uint4*)(wp);
        bf[s].q[1] = *(const uint4*)(wp + 8);
#pragma unroll
        for (int rt = 0; rt < 4; ++rt) {
            const unsigned short* xp = xrow + (size_t)rt * 16 * CC + c0;
            af[s][rt].q[0] = *(const uint4*)(xp);
            af[s][rt].q[1] = *(const uint4*)(xp + 8);
        }
    };

    const int NSTEP = CC / 32;                         // 24
    load_stage(0, 0);
    for (int s = 0; s < NSTEP; ++s) {
        int cur = s & 1, nxt = cur ^ 1;
        int c0n = (s + 1 < NSTEP) ? (s + 1) * 32 : 0;
        load_stage(nxt, c0n);
#pragma unroll
        for (int rt = 0; rt < 4; ++rt)
            acc[rt] = __builtin_amdgcn_wmma_f32_16x16x32_bf16(
                false, af[cur][rt].v, false, bf[cur].v, (short)0, acc[rt],
                false, false);
    }

    float bv = bias[o0 + ln];
#pragma unroll
    for (int rt = 0; rt < 4; ++rt)
#pragma unroll
        for (int i = 0; i < 8; ++i) {
            int trow = r0 + rt * 16 + (hi ? 8 + i : i);
            out[(size_t)trow * CC + o0 + ln] = acc[rt][i] + bv;
        }
}

// ---------------------------------------------------------------------------
extern "C" void kernel_launch(void* const* d_in, const int* in_sizes, int n_in,
                              void* d_out, int out_size, void* d_ws, size_t ws_size,
                              hipStream_t stream)
{
    const float* x  = (const float*)d_in[0];
    const float* pk = (const float*)d_in[1];
    const float* pv = (const float*)d_in[2];
    const float* wi = (const float*)d_in[3];   // in_proj_weight (3C, C)
    const float* wo = (const float*)d_in[4];   // out_proj_w (C, C)
    const float* bo = (const float*)d_in[5];   // out_proj_b (C,)
    float* out = (float*)d_out;

    // workspace carve-up (bf16 elements)
    unsigned short* ws = (unsigned short*)d_ws;
    const size_t xElems  = (size_t)NQ * BB * CC;        // 6,291,456
    const size_t wiElems = (size_t)3 * CC * CC;         // 1,769,472
    const size_t woElems = (size_t)CC * CC;             //   589,824
    const size_t qElems  = (size_t)BB * HH * NQ * HD;   // 6,291,456
    const size_t kElems  = (size_t)BB * HH * MPAD * HD; // 6,389,760
    unsigned short* xb    = ws;
    unsigned short* wib   = xb + xElems;
    unsigned short* wob   = wib + wiElems;
    unsigned short* qbuf  = wob + woElems;
    unsigned short* kbuf  = qbuf + qElems;
    unsigned short* vtbuf = kbuf + kElems;
    unsigned short* aout  = vtbuf + kElems;             // 6,291,456

    // Kernel 0: one-shot fp32 -> bf16 conversions
    cvt_bf16_kernel<<<(int)((xElems / 8 + 255) / 256), 256, 0, stream>>>(
        x, xb, (int)(xElems / 8));
    cvt_bf16_kernel<<<(int)((wiElems / 8 + 255) / 256), 256, 0, stream>>>(
        wi, wib, (int)(wiElems / 8));
    cvt_bf16_kernel<<<(int)((woElems / 8 + 255) / 256), 256, 0, stream>>>(
        wo, wob, (int)(woElems / 8));

    // Kernel 1: QKV projection.  waves = 128 rowgroups * 144 coltiles
    qkv_proj_kernel<<<(128 * 144) / 8, 256, 0, stream>>>(xb, wib, qbuf, kbuf, vtbuf);
    // Kernel 1b: prefix + padding
    prefix_pad_kernel<<<(BB * HH * HD * 32 + 255) / 256, 256, 0, stream>>>(
        pk, pv, kbuf, vtbuf);
    // Kernel 2: attention.  48 (b,h) * 16 row-blocks of 128 rows
    attn_kernel<<<48 * 16, 256, 0, stream>>>(qbuf, kbuf, vtbuf, aout);
    // Kernel 3: out projection.  waves = 128 rowgroups * 48 coltiles
    out_proj_kernel<<<(128 * 48) / 8, 256, 0, stream>>>(aout, wob, bo, out);
}
